// TemporalPatchDetokenizer_86947317940760
// MI455X (gfx1250) — matrix operand order repaired and checked by reading, not compiled, not run on hardware
//
#include <hip/hip_runtime.h>
#include <hip/hip_bf16.h>
#include <stddef.h>

// ---------------------------------------------------------------------------
// TemporalPatchDetokenizer for MI455X (gfx1250, wave32, WMMA).
//   patches = y @ W^T + b   (M=32704, N=1052, K=512)  -> bf16 WMMA, fp32 acc
//   out[b,jf,t] = (patches[t>>1, b, t&1, jf] + patches[(t>>1)-1, b, (t&1)+2, jf]) / count
// Workspace layout (needs ~139.3 MB):
//   [0 .. ROWS*OPAD*4)            fp32 patch buffer (padded N stride 1056)
//   [ROWS*OPAD*4 .. +PJF*DIM*2)   bf16 copy of W
// GEMM kernel stages the per-block 128-col W strip in LDS (130 KB dynamic).
// ---------------------------------------------------------------------------

typedef __attribute__((ext_vector_type(16))) __bf16         v16bf;
typedef __attribute__((ext_vector_type(8)))  unsigned short v8u;
typedef __attribute__((ext_vector_type(8)))  float          v8f;

#define NP    511
#define BS    64
#define DIM   512
#define JF    263
#define PJF   1052                  /* P*JF */
#define TT    1024
#define ROWS  (NP * BS)             /* 32704 = 2044 M-tiles of 16 */
#define OPAD  1056                  /* padded column stride (16B aligned rows) */
#define NTILES 8                    /* N tiles (of 16) per wave: 128 columns   */
#define NCOLS  (16 * NTILES)        /* 128 columns per block strip             */
#define KPAD   (DIM + 8)            /* LDS K-stride: 520 elems = 260 words     */
#define LDS_BYTES (NCOLS * KPAD * 2)/* 133,120 B < 320 KB WGP LDS              */
#define PATCH_ELEMS ((size_t)ROWS * OPAD)

// ---- kernel 1: W fp32 -> bf16 (row-major [o][d]) ---------------------------
__global__ __launch_bounds__(256)
void cvt_w_kernel(const float* __restrict__ W, __bf16* __restrict__ Wb) {
  int i = blockIdx.x * 256 + threadIdx.x;
  if (i < PJF * DIM) Wb[i] = (__bf16)W[i];   // v_cvt_pk_bf16_f32 (RNE)
}

// ---- kernel 2: GEMM via v_wmma_f32_16x16x32_bf16, B staged in LDS ----------
// One wave computes a 16(M) x 128(N) strip; 4 waves/block share the B strip.
// A (16x32 bf16) per-lane layout (ISA 7.12.2): lane<16: K = e (e<8), e+8 (e>=8)
//                                              lane>=16: K = e+8,     e+16
// B (32x16 bf16) per-lane layout: lane = N col, same K order as A: K = 8*(lane>>4)+e (e<8), 8*(lane>>4)+e+8 (e>=8).
__global__ __launch_bounds__(128)
void gemm_wmma_kernel(const float* __restrict__ Y,
                      const __bf16* __restrict__ Wb,
                      const float* __restrict__ bias,
                      float* __restrict__ patch) {
  extern __shared__ char smem[];
  __bf16* lbs = (__bf16*)smem;                     // [NCOLS][KPAD]

  const int tid   = threadIdx.x;
  const int lane  = tid & 31;
  const int wave  = tid >> 5;
  const int mTile = blockIdx.x * 4 + wave;         // 0..2043
  const int nBase = blockIdx.y * NCOLS;            // 0,128,...,1024
  const int lrow  = lane & 15;
  const int hi    = lane >> 4;                     // 0 or 1

  // ---- cooperative fill: B strip (128 cols x 512 K) global -> LDS ----------
  // thread t copies 16B chunks: threads 0-63 stream col c (coalesced 1KB row),
  // threads 64-127 stream col c+1; OOB columns clamp (never stored later).
  {
    const int q = tid & 63;                        // 16B chunk index along K
    for (int c = (tid >> 6); c < NCOLS; c += 2) {
      int gc = nBase + c;
      if (gc > PJF - 1) gc = PJF - 1;
      *(v8u*)(lbs + c * KPAD + q * 8) =
          *(const v8u*)(Wb + (size_t)gc * DIM + q * 8);
    }
  }
  __syncthreads();

  const float* yrow = Y + (size_t)(mTile * 16 + lrow) * DIM;
  const int akb = hi * 8;                          // A K-base per half-wave

  v8f acc[NTILES];
  #pragma unroll
  for (int j = 0; j < NTILES; ++j) acc[j] = (v8f)0.0f;

  // per-lane LDS base for each B tile fragment (bank-stride 4 -> conflict-min)
  const __bf16* lcol[NTILES];
  #pragma unroll
  for (int j = 0; j < NTILES; ++j)
    lcol[j] = lbs + (j * 16 + lrow) * KPAD + hi * 8;

  union CvtB { v8u h[2]; v16bf b; };

  #pragma unroll 2
  for (int k0 = 0; k0 < DIM; k0 += 32) {
    // ---- A fragment: two contiguous 32B chunks per lane (global) ----
    const float4 f0 = *(const float4*)(yrow + k0 + akb);
    const float4 f1 = *(const float4*)(yrow + k0 + akb + 4);
    const float4 f2 = *(const float4*)(yrow + k0 + 16 + akb);
    const float4 f3 = *(const float4*)(yrow + k0 + 16 + akb + 4);

    // ---- B fragments from LDS (2x ds_load_b128 per tile) ----
    CvtB B[NTILES];
    #pragma unroll
    for (int j = 0; j < NTILES; ++j) {
      B[j].h[0] = *(const v8u*)(lcol[j] + k0);
      B[j].h[1] = *(const v8u*)(lcol[j] + k0 + 16);
    }

    // ---- convert A while loads are in flight (v_cvt_pk_bf16_f32) ----
    v16bf ab;
    ab[0]  = (__bf16)f0.x; ab[1]  = (__bf16)f0.y; ab[2]  = (__bf16)f0.z; ab[3]  = (__bf16)f0.w;
    ab[4]  = (__bf16)f1.x; ab[5]  = (__bf16)f1.y; ab[6]  = (__bf16)f1.z; ab[7]  = (__bf16)f1.w;
    ab[8]  = (__bf16)f2.x; ab[9]  = (__bf16)f2.y; ab[10] = (__bf16)f2.z; ab[11] = (__bf16)f2.w;
    ab[12] = (__bf16)f3.x; ab[13] = (__bf16)f3.y; ab[14] = (__bf16)f3.z; ab[15] = (__bf16)f3.w;

    // ---- 8 WMMAs on the matrix pipe ----
    #pragma unroll
    for (int j = 0; j < NTILES; ++j)
      acc[j] = __builtin_amdgcn_wmma_f32_16x16x32_bf16(
                   false, ab, false, B[j].b, (short)0, acc[j], false, false);
  }

  // ---- store C tiles (+bias): lane = column, acc element v = row v + 8*hi --
  #pragma unroll
  for (int j = 0; j < NTILES; ++j) {
    const int o = nBase + j * 16 + lrow;
    if (o < PJF) {
      const float bv = bias[o];
      float* base = patch + (size_t)(mTile * 16 + hi * 8) * OPAD + o;
      #pragma unroll
      for (int v = 0; v < 8; ++v)
        base[(size_t)v * OPAD] = acc[j][v] + bv;
    }
  }
}

// ---- kernel 3: closed-form overlap-add + normalization ---------------------
// out[b,jf,t] = (patch[n0,b,p0*JF+jf] (n0<NP) + patch[n0-1,b,(p0+2)*JF+jf] (n0>0)) / count
__global__ __launch_bounds__(256)
void scatter_norm_kernel(const float* __restrict__ patch, float* __restrict__ out) {
  int idx = blockIdx.x * 256 + threadIdx.x;
  if (idx >= BS * JF * TT) return;
  int t   = idx & (TT - 1);
  int rem = idx >> 10;            // b*JF + jf
  int jf  = rem % JF;
  int b   = rem / JF;
  int p0  = t & 1;
  int n0  = t >> 1;

  float s = 0.0f;
  int   c = 0;
  if (n0 < NP) { s += patch[(size_t)(n0 * BS + b) * OPAD + p0 * JF + jf];             ++c; }
  if (n0 >= 1) { s += patch[(size_t)((n0 - 1) * BS + b) * OPAD + (p0 + 2) * JF + jf]; ++c; }
  out[idx] = (c == 2) ? s * 0.5f : s;   // out is (b, jf, t) flat == idx
}

// ---------------------------------------------------------------------------
extern "C" void kernel_launch(void* const* d_in, const int* in_sizes, int n_in,
                              void* d_out, int out_size, void* d_ws, size_t ws_size,
                              hipStream_t stream) {
  const float* Y    = (const float*)d_in[0];   // (511, 64, 512) fp32
  const float* W    = (const float*)d_in[1];   // (1052, 512) fp32
  const float* bias = (const float*)d_in[2];   // (1052,) fp32
  // d_in[3] = starts (= 2n), d_in[4] = T (1024), d_in[5] = P (4): baked in.

  float*  patch = (float*)d_ws;
  __bf16* Wb    = (__bf16*)((char*)d_ws + PATCH_ELEMS * sizeof(float));

  cvt_w_kernel<<<(PJF * DIM) / 256, 256, 0, stream>>>(W, Wb);
  gemm_wmma_kernel<<<dim3(ROWS / 64, 9), 128, LDS_BYTES, stream>>>(Y, Wb, bias, patch);
  scatter_norm_kernel<<<(BS * JF * TT) / 256, 256, 0, stream>>>(patch, (float*)d_out);
}
